// BiMatchLSTM_52201032516071
// MI455X (gfx1250) — matrix operand order repaired and checked by reading, not compiled, not run on hardware
//
#include <hip/hip_runtime.h>
#include <cstdint>
#include <cstddef>

// ---------------------------------------------------------------------------
// Types / constants
// ---------------------------------------------------------------------------
typedef __attribute__((ext_vector_type(16))) __bf16 v16bf;
typedef __attribute__((ext_vector_type(8)))  float  v8f;

#define DEVINL static __device__ __forceinline__

constexpr int BB   = 64;    // batch
constexpr int TPP  = 400;   // p time steps
constexpr int TQQ  = 60;    // q time steps
constexpr int DPP  = 512;   // p feature dim
constexpr int DQQ  = 512;   // q feature dim
constexpr int NHH  = 256;   // hidden
constexpr int NGG  = 1024;  // 4*NH gate columns
constexpr int KGG  = 1280;  // DP + DQ + NH  (fused gate GEMM K)
constexpr int NWG  = 8;     // workgroups per direction
constexpr int SLICE = NHH / NWG;   // 32 hidden units per WG (== one WMMA K chunk)
constexpr int NTHREADS = 256;      // 8 waves (wave32)
constexpr int BPG  = 4;            // B k-chunks staged per async group
constexpr int NGRP = (KGG / 32) / BPG;   // 10 groups
constexpr int PARTSZ = NWG * BB * NHH;   // hWr partial buffer per (dir,parity)

union Frag { v16bf v; unsigned short us[16]; uint4 q2[2]; };

// ---------------------------------------------------------------------------
// Small helpers
// ---------------------------------------------------------------------------
DEVINL unsigned short f2bf(float f) {
  unsigned u = __builtin_bit_cast(unsigned, f);
  u += 0x7FFFu + ((u >> 16) & 1u);          // round-to-nearest-even
  return (unsigned short)(u >> 16);
}

DEVINL float ftanh(float x) {
  float ax = __builtin_fabsf(x);
  float e  = __expf(-2.0f * ax);            // v_exp_f32 path
  float r  = (1.0f - e) * __builtin_amdgcn_rcpf(1.0f + e);
  return __builtin_copysignf(r, x);
}

DEVINL float fsigm(float x) {
  return __builtin_amdgcn_rcpf(1.0f + __expf(-x));
}

DEVINL v8f wmma_bf(v16bf a, v16bf b, v8f c) {
  return __builtin_amdgcn_wmma_f32_16x16x32_bf16(false, a, false, b,
                                                 (short)0, c, false, false);
}

// B fragment from pre-swizzled weights: layout [K/32][N/16][lane(32)][e(16)]
DEVINL v16bf load_bsw(const unsigned short* Wsw, int Ntiles, int kt, int ntg, int lane) {
  const unsigned short* src = Wsw + ((((size_t)kt * Ntiles + ntg) * 32 + lane) << 4);
  Frag f;
  f.q2[0] = *(const uint4*)(src);
  f.q2[1] = *(const uint4*)(src + 8);
  return f.v;
}

// B fragment from LDS-staged swizzled data (16 contiguous bf16 per lane)
DEVINL v16bf load_blds(const unsigned short* src) {
  Frag f;
  f.q2[0] = *(const uint4*)(src);
  f.q2[1] = *(const uint4*)(src + 8);
  return f.v;
}

// A fragment from row-major bf16; src pre-offset to rowBase + 8*(lane>>4)
DEVINL v16bf load_abf(const unsigned short* src) {
  Frag f;
  f.q2[0] = *(const uint4*)(src);
  f.q2[1] = *(const uint4*)(src + 16);
  return f.v;
}

// A fragment from row-major f32 (converted on the fly); src pre-offset
DEVINL v16bf load_af32(const float* src) {
  float4 a = *(const float4*)(src);
  float4 b = *(const float4*)(src + 4);
  float4 c = *(const float4*)(src + 16);
  float4 d = *(const float4*)(src + 20);
  Frag f;
  f.us[0]  = f2bf(a.x); f.us[1]  = f2bf(a.y); f.us[2]  = f2bf(a.z); f.us[3]  = f2bf(a.w);
  f.us[4]  = f2bf(b.x); f.us[5]  = f2bf(b.y); f.us[6]  = f2bf(b.z); f.us[7]  = f2bf(b.w);
  f.us[8]  = f2bf(c.x); f.us[9]  = f2bf(c.y); f.us[10] = f2bf(c.z); f.us[11] = f2bf(c.w);
  f.us[12] = f2bf(d.x); f.us[13] = f2bf(d.y); f.us[14] = f2bf(d.z); f.us[15] = f2bf(d.w);
  return f.v;
}

// CDNA5 async global->LDS copy (ASYNCcnt tracked); vdst VGPR = LDS byte offset
DEVINL void async_load_b128(const unsigned short* gsrc, unsigned lds_off) {
  asm volatile("global_load_async_to_lds_b128 %0, %1, off"
               :: "v"(lds_off), "v"(gsrc) : "memory");
}
DEVINL void wait_async_le8() { asm volatile("s_wait_asynccnt 0x8" ::: "memory"); }
DEVINL void wait_async_0()   { asm volatile("s_wait_asynccnt 0x0" ::: "memory"); }

// Grid-wide barrier (per-direction)
DEVINL void gbar(unsigned* cnt, unsigned* gen, unsigned n) {
  __syncthreads();
  if (threadIdx.x == 0) {
    __threadfence();
    unsigned g = __hip_atomic_load(gen, __ATOMIC_RELAXED, __HIP_MEMORY_SCOPE_AGENT);
    unsigned a = __hip_atomic_fetch_add(cnt, 1u, __ATOMIC_ACQ_REL, __HIP_MEMORY_SCOPE_AGENT);
    if (a == n - 1u) {
      __hip_atomic_store(cnt, 0u, __ATOMIC_RELAXED, __HIP_MEMORY_SCOPE_AGENT);
      __hip_atomic_store(gen, g + 1u, __ATOMIC_RELEASE, __HIP_MEMORY_SCOPE_AGENT);
    } else {
      while (__hip_atomic_load(gen, __ATOMIC_ACQUIRE, __HIP_MEMORY_SCOPE_AGENT) == g)
        __builtin_amdgcn_s_sleep(2);
    }
  }
  __syncthreads();
}

// ---------------------------------------------------------------------------
// Utility kernels
// ---------------------------------------------------------------------------
__global__ void k_zero(float* p, long n) {
  long i = (long)blockIdx.x * blockDim.x + threadIdx.x;
  if (i < n) p[i] = 0.0f;
}

__global__ void k_tobf(const float* in, unsigned short* out, long n) {
  long i = (long)blockIdx.x * blockDim.x + threadIdx.x;
  if (i < n) out[i] = f2bf(in[i]);
}

// Row-major f32 W[K,N] -> swizzled bf16 B-fragment order
__global__ void k_swz(const float* W, int ldw, unsigned short* out, int K, int N) {
  long t = (long)blockIdx.x * blockDim.x + threadIdx.x;
  long total = (long)K * N;
  if (t >= total) return;
  int e   = (int)(t & 15);
  int L   = (int)((t >> 4) & 31);
  long rest = t >> 9;
  int Ntg = N >> 4;
  int ntg = (int)(rest % Ntg);
  int kt  = (int)(rest / Ntg);
  int k = kt * 32 + ((L >> 4) << 3) + (e & 7) + ((e >> 3) << 4);
  int n = ntg * 16 + (L & 15);
  out[t] = f2bf(W[(size_t)k * ldw + n]);
}

// C[M,N] = A[M,K](f32) @ Bsw(swizzled bf16) + bias ; one wave = 16x64 tile
__global__ void k_gemm(const float* A, int K, const unsigned short* Bsw,
                       const float* bias, float* C, int N, int tiles) {
  int wid  = (blockIdx.x * blockDim.x + threadIdx.x) >> 5;
  int lane = threadIdx.x & 31;
  if (wid >= tiles) return;
  int ngroups = N >> 6;
  int mt = wid / ngroups;
  int ng = wid % ngroups;
  int Nt = N >> 4;

  v8f z = {0.f,0.f,0.f,0.f,0.f,0.f,0.f,0.f};
  v8f acc[4] = {z, z, z, z};

  int row = mt * 16 + (lane & 15);
  const float* abase = A + (size_t)row * K + 8 * (lane >> 4);
  int kchunks = K >> 5;
  for (int kt = 0; kt < kchunks; ++kt) {
    v16bf a = load_af32(abase + kt * 32);
#pragma unroll
    for (int i = 0; i < 4; ++i)
      acc[i] = wmma_bf(a, load_bsw(Bsw, Nt, kt, ng * 4 + i, lane), acc[i]);
  }
#pragma unroll
  for (int i = 0; i < 4; ++i) {
    int col = (ng * 4 + i) * 16 + (lane & 15);
    float bi = bias ? bias[col] : 0.0f;
#pragma unroll
    for (int r = 0; r < 8; ++r) {
      int rr = mt * 16 + r + 8 * (lane >> 4);
      C[(size_t)rr * N + col] = acc[i][r] + bi;
    }
  }
}

// ---------------------------------------------------------------------------
// Recurrent (persistent, cooperative) kernel — 2 grid barriers per step
// ---------------------------------------------------------------------------
struct Ptrs {
  const float *q, *mask_p, *mask_q, *w_score, *bias_f, *bias_b;
  const float *pWp, *qWq;
  const unsigned short *p_bf, *Wr_sw, *Wg_f, *Wg_b;
  float *h, *c;                  // [2][B*NH]
  float *part;                   // [2 dir][2 parity][NWG][B][NH] hWr partials
  unsigned short *hbf, *ctxbf;   // hbf: [2 dir][2 parity][B*NH]
  unsigned *bar;
  float *out_states, *out_last;
};

__global__ __launch_bounds__(NTHREADS)
void k_recurrent(Ptrs P) {
  const int dir  = blockIdx.x / NWG;
  const int wg   = blockIdx.x % NWG;
  const int tid  = threadIdx.x;
  const int wave = tid >> 5;
  const int lane = tid & 31;
  const int j0   = wg * SLICE;
  const int b0   = wg * (BB / NWG);      // 8 owned batch rows for attention

  float* h = P.h + (size_t)dir * BB * NHH;
  float* c = P.c + (size_t)dir * BB * NHH;
  unsigned short* ctxbf = P.ctxbf + (size_t)dir * BB * DQQ;
  const unsigned short* Wg   = dir ? P.Wg_b   : P.Wg_f;
  const float*          bias = dir ? P.bias_b : P.bias_f;
  unsigned* cnt = P.bar + dir * 32;
  unsigned* gen = cnt + 16;

  __shared__ unsigned short ldsB[2][BPG * 8 * 512];  // 64 KB async B staging
  __shared__ float          lds_g[4][BB][SLICE];     // 32 KB gate tiles
  __shared__ float          lds_hwr[8][NHH];         //  8 KB reduced hWr rows
  __shared__ float          lds_s[8][64];            //  2 KB scores
  __shared__ unsigned short lds_hn[BB][SLICE];       //  4 KB new-h slice (bf16)

  const unsigned ldsB_base = (unsigned)(uintptr_t)(&ldsB[0][0]);

  for (int t = 0; t < TPP; ++t) {
    const int tp = dir ? (TPP - 1 - t) : t;
    const unsigned short* hbf_r = P.hbf + ((size_t)(dir * 2 + (t & 1))) * BB * NHH;
    unsigned short*       hbf_w = P.hbf + ((size_t)(dir * 2 + ((t + 1) & 1))) * BB * NHH;
    const float* partR = P.part + ((size_t)(dir * 2 + (t & 1))) * PARTSZ;
    float*       partW = P.part + ((size_t)(dir * 2 + ((t + 1) & 1))) * PARTSZ;

    // ================= Phase BC: attention for owned 8 batch rows ==========
    // 1) reduce hWr partials into LDS
#pragma unroll
    for (int r = 0; r < 8; ++r) {
      int out = tid + NTHREADS * r;          // 0..2047 = 8 b x 256 n
      int bl = out >> 8, n = out & 255;
      float acc = 0.0f;
#pragma unroll
      for (int w = 0; w < NWG; ++w)
        acc += partR[((size_t)w * BB + (b0 + bl)) * NHH + n];
      lds_hwr[bl][n] = acc;
    }
    __syncthreads();

    // 2) scores s[bl, tq] = sum_h tanh(qWq + pWp + hWr) * w_score + mask bias
    for (int pp = 0; pp < 2; ++pp) {
      int pair = tid + NTHREADS * pp;
      if (pair < 8 * TQQ) {
        int bl = pair / TQQ, tq = pair % TQQ;
        int b = b0 + bl;
        const float* qr = P.qWq + (size_t)(b * TQQ + tq) * NHH;
        const float* pr = P.pWp + (size_t)(b * TPP + tp) * NHH;
        const float* hr = &lds_hwr[bl][0];
        float acc = 0.0f;
        for (int hh = 0; hh < NHH; hh += 4) {
          float4 qa = *(const float4*)(qr + hh);
          float4 pa = *(const float4*)(pr + hh);
          float4 ha = *(const float4*)(hr + hh);
          float4 wv = *(const float4*)(P.w_score + hh);
          acc += ftanh(qa.x + pa.x + ha.x) * wv.x;
          acc += ftanh(qa.y + pa.y + ha.y) * wv.y;
          acc += ftanh(qa.z + pa.z + ha.z) * wv.z;
          acc += ftanh(qa.w + pa.w + ha.w) * wv.w;
        }
        acc += (P.mask_q[b * TQQ + tq] - 1.0f) * 1000000000.0f;
        lds_s[bl][tq] = acc;
      }
    }
    __syncthreads();

    // 3) softmax + ctx = alpha @ q  (32 threads per b, 16 dims each)
    {
      int bl = tid >> 5;
      int d0 = (tid & 31) * 16;
      int b  = b0 + bl;
      float mx = -3.4e38f;
      for (int tq = 0; tq < TQQ; ++tq) mx = fmaxf(mx, lds_s[bl][tq]);
      float sum = 0.0f;
      for (int tq = 0; tq < TQQ; ++tq) sum += __expf(lds_s[bl][tq] - mx);
      float inv = __builtin_amdgcn_rcpf(sum);
      float accv[16];
#pragma unroll
      for (int i = 0; i < 16; ++i) accv[i] = 0.0f;
      for (int tq = 0; tq < TQQ; ++tq) {
        float a = __expf(lds_s[bl][tq] - mx) * inv;
        const float* qrow = P.q + (size_t)(b * TQQ + tq) * DQQ + d0;
#pragma unroll
        for (int i = 0; i < 16; i += 4) {
          float4 qv = *(const float4*)(qrow + i);
          accv[i + 0] += a * qv.x; accv[i + 1] += a * qv.y;
          accv[i + 2] += a * qv.z; accv[i + 3] += a * qv.w;
        }
      }
#pragma unroll
      for (int i = 0; i < 16; ++i) ctxbf[(size_t)b * DQQ + d0 + i] = f2bf(accv[i]);
    }
    gbar(cnt, gen, NWG);

    // ================= Phase D: gates GEMM + LSTM + hWr partial ============
    {
      int mt = wave & 3, half = wave >> 2;
      v8f z = {0.f,0.f,0.f,0.f,0.f,0.f,0.f,0.f};
      v8f acc[4] = {z, z, z, z};
      int brow = mt * 16 + (lane & 15);
      const unsigned short* pbase = P.p_bf + ((size_t)(brow * TPP + tp)) * DPP + 8 * (lane >> 4);
      const unsigned short* cbase = ctxbf + (size_t)brow * DQQ + 8 * (lane >> 4);
      const unsigned short* hbase = hbf_r + (size_t)brow * NHH + 8 * (lane >> 4);

      // async issue of one group (BPG k-chunks x 8 N-tiles) into ldsB[g&1]
      auto issue_group = [&](int g) {
        unsigned pbaseB = ldsB_base + (unsigned)((g & 1) * (BPG * 8 * 512 * 2));
#pragma unroll
        for (int r = 0; r < 8; ++r) {
          int u = tid + NTHREADS * r;        // 0..2047 -> 32KB / 16B
          int ktl  = u >> 9;                 // 0..3
          int rem  = u & 511;
          int tile = rem >> 6;               // 0..7 (= ntl)
          int prt  = rem & 63;
          int ll = prt >> 1, hf = prt & 1;
          int gg = tile >> 1, sub = tile & 1;
          int ntg = gg * (NHH / 16) + (j0 >> 4) + sub;
          const unsigned short* src =
              Wg + (((size_t)((g * BPG + ktl) * (NGG / 16) + ntg)) * 32 + ll) * 16 + hf * 8;
          unsigned dst = pbaseB + (unsigned)(ktl * 8192 + tile * 1024 + ll * 32 + hf * 16);
          async_load_b128(src, dst);
        }
      };

      issue_group(0);
      for (int g = 0; g < NGRP; ++g) {
        if (g + 1 < NGRP) { issue_group(g + 1); wait_async_le8(); }
        else              { wait_async_0(); }
        __syncthreads();                     // group g staged for all waves
        const unsigned short* bbuf = &ldsB[g & 1][0];
#pragma unroll
        for (int ktl = 0; ktl < BPG; ++ktl) {
          int kt = g * BPG + ktl;
          const unsigned short* asrc;
          if (kt < 16)      asrc = pbase + kt * 32;
          else if (kt < 32) asrc = cbase + (kt - 16) * 32;
          else              asrc = hbase + (kt - 32) * 32;
          v16bf a = load_abf(asrc);
#pragma unroll
          for (int i = 0; i < 4; ++i) {
            int ntl = half * 4 + i;
            acc[i] = wmma_bf(a, load_blds(bbuf + ktl * 4096 + ntl * 512 + lane * 16), acc[i]);
          }
        }
        __syncthreads();                     // done reading before buffer reuse
      }
#pragma unroll
      for (int i = 0; i < 4; ++i) {
        int ntl = half * 4 + i;
        int gg = ntl >> 1, sub = ntl & 1;
        int coln = gg * NHH + j0 + sub * 16 + (lane & 15);
        float bi = bias[coln];
#pragma unroll
        for (int r = 0; r < 8; ++r) {
          int rr = mt * 16 + r + 8 * (lane >> 4);
          lds_g[gg][rr][sub * 16 + (lane & 15)] = acc[i][r] + bi;
        }
      }
    }
    __syncthreads();

    // pointwise LSTM update for owned hidden slice
    for (int u = 0; u < (BB * SLICE) / NTHREADS; ++u) {
      int idx = tid + NTHREADS * u;          // 0..2047
      int b  = idx >> 5;
      int jj = idx & 31;
      int j  = j0 + jj;
      float gi = lds_g[0][b][jj];
      float gf = lds_g[1][b][jj];
      float gg = lds_g[2][b][jj];
      float go = lds_g[3][b][jj];
      float i_ = fsigm(gi), f_ = fsigm(gf), o_ = fsigm(go);
      float g_ = ftanh(gg);
      size_t hidx = (size_t)b * NHH + j;
      float cold = c[hidx], hold = h[hidx];
      float nc = f_ * cold + i_ * g_;
      float nh = o_ * ftanh(nc);
      float m = P.mask_p[b * TPP + tp];
      nh = nh * m + hold * (1.0f - m);
      nc = nc * m + cold * (1.0f - m);
      c[hidx] = nc;
      h[hidx] = nh;
      unsigned short nb = f2bf(nh);
      hbf_w[hidx] = nb;
      lds_hn[b][jj] = nb;
      P.out_states[((size_t)b * TPP + tp) * (2 * NHH) + dir * NHH + j] = nh * m;
      if (t == TPP - 1)
        P.out_last[(size_t)b * (2 * NHH) + dir * NHH + j] = nh;
    }
    __syncthreads();

    // hWr partial: h_new[:, j0:j0+32] @ Wr[j0:j0+32, :]  (K = one chunk = wg)
    {
      int mt = wave & 3;
      int ng0 = (wave >> 2) * 8;             // 8 of 16 N-tiles
      int brow = mt * 16 + (lane & 15);
      v16bf a = load_abf(&lds_hn[brow][0] + 8 * (lane >> 4));
      float* pw = partW + (size_t)wg * BB * NHH;
      v8f z = {0.f,0.f,0.f,0.f,0.f,0.f,0.f,0.f};
#pragma unroll
      for (int i = 0; i < 8; ++i) {
        int ntg = ng0 + i;
        v8f pc = wmma_bf(a, load_bsw(P.Wr_sw, NHH / 16, wg, ntg, lane), z);
#pragma unroll
        for (int r = 0; r < 8; ++r) {
          int rr = mt * 16 + r + 8 * (lane >> 4);
          pw[(size_t)rr * NHH + ntg * 16 + (lane & 15)] = pc[r];
        }
      }
    }
    gbar(cnt, gen, NWG);   // new h / partials visible to all WGs
  }
}

// ---------------------------------------------------------------------------
// Host launch
// ---------------------------------------------------------------------------
static inline int divup(long a, long b) { return (int)((a + b - 1) / b); }

extern "C" void kernel_launch(void* const* d_in, const int* in_sizes, int n_in,
                              void* d_out, int out_size, void* d_ws, size_t ws_size,
                              hipStream_t stream) {
  (void)in_sizes; (void)n_in; (void)out_size; (void)ws_size;

  const float* p       = (const float*)d_in[0];
  const float* mask_p  = (const float*)d_in[1];
  const float* q       = (const float*)d_in[2];
  const float* mask_q  = (const float*)d_in[3];
  const float* Wp      = (const float*)d_in[4];
  const float* Wq      = (const float*)d_in[5];
  const float* Wr      = (const float*)d_in[6];
  const float* b_att   = (const float*)d_in[7];
  const float* w_score = (const float*)d_in[8];
  const float* Wih_f   = (const float*)d_in[9];
  const float* Whh_f   = (const float*)d_in[10];
  const float* b_f     = (const float*)d_in[11];
  const float* Wih_b   = (const float*)d_in[12];
  const float* Whh_b   = (const float*)d_in[13];
  const float* b_b     = (const float*)d_in[14];

  char* base = (char*)d_ws;
  auto take = [&](size_t bytes) -> char* {
    char* r = base;
    base += (bytes + 255) & ~(size_t)255;
    return r;
  };

  float*          pWp   = (float*)take((size_t)BB * TPP * NHH * 4);
  float*          qWq   = (float*)take((size_t)BB * TQQ * NHH * 4);
  unsigned short* p_bf  = (unsigned short*)take((size_t)BB * TPP * DPP * 2);
  unsigned short* Wp_sw = (unsigned short*)take((size_t)DPP * NHH * 2);
  unsigned short* Wq_sw = (unsigned short*)take((size_t)DQQ * NHH * 2);
  unsigned short* Wr_sw = (unsigned short*)take((size_t)NHH * NHH * 2);
  unsigned short* Wg_fs = (unsigned short*)take((size_t)KGG * NGG * 2);
  unsigned short* Wg_bs = (unsigned short*)take((size_t)KGG * NGG * 2);

  char* stateStart = base;
  float*          hbuf  = (float*)take((size_t)2 * BB * NHH * 4);
  float*          cbuf  = (float*)take((size_t)2 * BB * NHH * 4);
  float*          partb = (float*)take((size_t)2 * 2 * PARTSZ * 4);   // dir x parity
  unsigned short* ctxbf = (unsigned short*)take((size_t)2 * BB * DQQ * 2);
  unsigned short* hbf   = (unsigned short*)take((size_t)4 * BB * NHH * 2); // dir x parity
  unsigned*       bar   = (unsigned*)take(256);
  long stateFloats = (long)(base - stateStart) / 4;

  // 1) zero persistent state + barrier
  k_zero<<<divup(stateFloats, 256), 256, 0, stream>>>((float*)stateStart, stateFloats);

  // 2) weight conversion / swizzling to bf16 B-fragment order
  k_swz<<<divup((long)DPP * NHH, 256), 256, 0, stream>>>(Wp, NHH, Wp_sw, DPP, NHH);
  k_swz<<<divup((long)DQQ * NHH, 256), 256, 0, stream>>>(Wq, NHH, Wq_sw, DQQ, NHH);
  k_swz<<<divup((long)NHH * NHH, 256), 256, 0, stream>>>(Wr, NHH, Wr_sw, NHH, NHH);
  // fused gate weight: rows [0,512)=Wih_p, [512,1024)=Wih_ctx, [1024,1280)=Whh
  k_swz<<<divup((long)512 * NGG, 256), 256, 0, stream>>>(Wih_f,             NGG, Wg_fs,              512, NGG);
  k_swz<<<divup((long)512 * NGG, 256), 256, 0, stream>>>(Wih_f + 512 * NGG, NGG, Wg_fs + 512 * NGG,  512, NGG);
  k_swz<<<divup((long)256 * NGG, 256), 256, 0, stream>>>(Whh_f,             NGG, Wg_fs + 1024 * NGG, 256, NGG);
  k_swz<<<divup((long)512 * NGG, 256), 256, 0, stream>>>(Wih_b,             NGG, Wg_bs,              512, NGG);
  k_swz<<<divup((long)512 * NGG, 256), 256, 0, stream>>>(Wih_b + 512 * NGG, NGG, Wg_bs + 512 * NGG,  512, NGG);
  k_swz<<<divup((long)256 * NGG, 256), 256, 0, stream>>>(Whh_b,             NGG, Wg_bs + 1024 * NGG, 256, NGG);

  // 3) p -> bf16 (A-side of fused gate GEMM)
  k_tobf<<<divup((long)BB * TPP * DPP, 256), 256, 0, stream>>>(p, p_bf, (long)BB * TPP * DPP);

  // 4) precompute pWp = p @ Wp ; qWq = q @ Wq + b_att  (WMMA GEMMs)
  {
    int tiles = (BB * TPP / 16) * (NHH / 64);
    k_gemm<<<divup((long)tiles * 32, 256), 256, 0, stream>>>(p, DPP, Wp_sw, nullptr, pWp, NHH, tiles);
  }
  {
    int tiles = (BB * TQQ / 16) * (NHH / 64);
    k_gemm<<<divup((long)tiles * 32, 256), 256, 0, stream>>>(q, DQQ, Wq_sw, b_att, qWq, NHH, tiles);
  }

  // 5) cooperative bidirectional recurrence
  Ptrs P;
  P.q = q; P.mask_p = mask_p; P.mask_q = mask_q; P.w_score = w_score;
  P.bias_f = b_f; P.bias_b = b_b;
  P.pWp = pWp; P.qWq = qWq;
  P.p_bf = p_bf; P.Wr_sw = Wr_sw; P.Wg_f = Wg_fs; P.Wg_b = Wg_bs;
  P.h = hbuf; P.c = cbuf; P.part = partb;
  P.hbf = hbf; P.ctxbf = ctxbf; P.bar = bar;
  P.out_states = (float*)d_out;
  P.out_last   = (float*)d_out + (size_t)BB * TPP * (2 * NHH);

  k_recurrent<<<2 * NWG, NTHREADS, 0, stream>>>(P);
}
